// ThinkingLSTM_29686813950705
// MI455X (gfx1250) — matrix operand order repaired and checked by reading, not compiled, not run on hardware
//
#include <hip/hip_runtime.h>
#include <hip/hip_bf16.h>
#include <stdint.h>

// ---------------------------------------------------------------------------
// ThinkingLSTM for MI455X (gfx1250): bf16 WMMA GEMMs + elementwise cell/ctrl.
// B=256, T_ENC=128, VOCAB=1024, HID=1024, 4H=4096, MAX_THINK=16, N=125.
// A-tiles staged to LDS with GLOBAL_LOAD_ASYNC_TO_LDS_B64 (ASYNCcnt path).
// ---------------------------------------------------------------------------

typedef __bf16 bf16_t;
typedef __attribute__((ext_vector_type(16))) __bf16 v16bf;
typedef __attribute__((ext_vector_type(8)))  __bf16 v8bf;
typedef __attribute__((ext_vector_type(8)))  float  v8f;

#define BSZ        256
#define T_ENC      128
#define VOCAB      1024
#define HID        1024
#define GATES4H    4096
#define KCAT       2048
#define MAX_THINK  16
#define N_OUT      125
#define T_DEC      (MAX_THINK + N_OUT)   // 141
#define NUM_CLASSES 1023

// ===========================================================================
// WMMA GEMM:  C[M,N] = A[M,K](bf16,row-major) * W[N,K](bf16,row-major)^T + bias
// Block = 256 threads = 8 waves. Block tile = 32(M) x 512(N).
// Wave tile = 32(M) x 64(N)  -> 2 x 4 accumulators of v8f.
// A tile (32x32 bf16) staged through LDS via async load, double buffered.
// ===========================================================================
__global__ __launch_bounds__(256) void gemm_bf16_wmma(
    const bf16_t* __restrict__ A, int lda,
    const bf16_t* __restrict__ W, int ldb,
    const float*  __restrict__ bias,
    float* __restrict__ C, int ldc,
    int K)
{
    __shared__ bf16_t sA[2][32 * 32];

    const int tid  = threadIdx.x;
    const int wave = tid >> 5;
    const int lane = tid & 31;
    const int m0   = blockIdx.y * 32;
    const int nw   = blockIdx.x * 512 + wave * 64;

    v8f acc[2][4];
    const v8f vzero = {0.f, 0.f, 0.f, 0.f, 0.f, 0.f, 0.f, 0.f};
#pragma unroll
    for (int i = 0; i < 2; ++i)
#pragma unroll
        for (int j = 0; j < 4; ++j) acc[i][j] = vzero;

    const int nK  = K >> 5;
    const int row = tid >> 3;        // 0..31
    const int seg = tid & 7;         // 0..7  (8 bytes = 4 bf16 each)

    // prologue: async-stage k-chunk 0 into buffer 0 (memory -> LDS, no VGPRs)
    {
        uint32_t ldst = (uint32_t)(uintptr_t)&sA[0][row * 32 + seg * 4];
        uint64_t ga   = (uint64_t)(uintptr_t)(A + (size_t)(m0 + row) * lda + seg * 4);
        asm volatile("global_load_async_to_lds_b64 %0, %1, off"
                     :: "v"(ldst), "v"(ga) : "memory");
    }

    const int half = lane >> 4;      // 0/1 : K sub-stripe select
    const int rc   = lane & 15;      // row (A frag) / column (B frag, D col)

    int buf = 0;
    for (int kk = 0; kk < nK; ++kk) {
        // my async LDS writes from the previous issue must land, then sync WG
        asm volatile("s_wait_asynccnt 0x0" ::: "memory");
        __syncthreads();

        // async-prefetch next A tile into the other buffer (overlaps WMMAs)
        if (kk + 1 < nK) {
            uint32_t ldst = (uint32_t)(uintptr_t)&sA[buf ^ 1][row * 32 + seg * 4];
            uint64_t ga   = (uint64_t)(uintptr_t)(A + (size_t)(m0 + row) * lda +
                                                  (kk + 1) * 32 + seg * 4);
            asm volatile("global_load_async_to_lds_b64 %0, %1, off"
                         :: "v"(ldst), "v"(ga) : "memory");
        }

        // ---- A fragments from LDS (16x32 bf16 layout, ISA 7.12.2) ----
        v16bf afrag[2];
#pragma unroll
        for (int i = 0; i < 2; ++i) {
            const bf16_t* base = &sA[buf][(i * 16 + rc) * 32];
            union { v16bf v; v8bf h[2]; } ua;
            ua.h[0] = *(const v8bf*)(base + half * 8);        // K = h*8 .. h*8+7
            ua.h[1] = *(const v8bf*)(base + 16 + half * 8);   // K = 16+h*8 ..
            afrag[i] = ua.v;
        }

        // ---- B fragments from global (32x16 bf16: lane n = col n,
        //      K = h*8 .. h*8+7, then 16+h*8 .. 16+h*8+7) ----
        const int k0 = kk * 32;
#pragma unroll
        for (int j = 0; j < 4; ++j) {
            const int ncol = nw + j * 16 + rc;
            const bf16_t* wp = W + (size_t)ncol * ldb + k0;
            __builtin_prefetch(wp + 32, 0, 3);   // near-scope prefetch, next chunk
            union { v16bf v; v8bf h[2]; } ub;
            ub.h[0] = *(const v8bf*)(wp + half * 8);
            ub.h[1] = *(const v8bf*)(wp + 16 + half * 8);
            acc[0][j] = __builtin_amdgcn_wmma_f32_16x16x32_bf16(
                false, afrag[0], false, ub.v, (short)0, acc[0][j], false, false);
            acc[1][j] = __builtin_amdgcn_wmma_f32_16x16x32_bf16(
                false, afrag[1], false, ub.v, (short)0, acc[1][j], false, false);
        }
        buf ^= 1;
    }

    // ---- epilogue: D layout row = half*8 + q, col = rc ----
#pragma unroll
    for (int i = 0; i < 2; ++i) {
#pragma unroll
        for (int j = 0; j < 4; ++j) {
            const int nc = nw + j * 16 + rc;
            const float bv = bias ? bias[nc] : 0.f;
#pragma unroll
            for (int q = 0; q < 8; ++q) {
                const int r = m0 + i * 16 + half * 8 + q;
                C[(size_t)r * ldc + nc] = acc[i][j][q] + bv;
            }
        }
    }
}

// ===========================================================================
// Elementwise / control kernels
// ===========================================================================
__device__ __forceinline__ float sig_f(float x) { return 1.f / (1.f + expf(-x)); }

__device__ __forceinline__ uint64_t splitmix64(uint64_t x) {
    x += 0x9E3779B97F4A7C15ULL;
    x = (x ^ (x >> 30)) * 0xBF58476D1CE4E5B9ULL;
    x = (x ^ (x >> 27)) * 0x94D049BB133111EBULL;
    return x ^ (x >> 31);
}

__device__ __forceinline__ float gumbel_f(int t, int b, int v) {
    uint64_t s = ((uint64_t)(t + 1) << 42) ^ ((uint64_t)(b + 1) << 21) ^
                 (uint64_t)v ^ 0x2A2A2A2AULL;   // seed 42 flavored
    s = splitmix64(splitmix64(s));
    float u = (float)(s >> 40) * (1.0f / 16777216.0f);      // [0,1)
    u = fminf(fmaxf(u, 1e-12f), 0.99999994f);
    return -logf(-logf(u));
}

__global__ void fill_zero_u32(uint32_t* p, size_t n) {
    size_t i = (size_t)blockIdx.x * blockDim.x + threadIdx.x;
    size_t st = (size_t)gridDim.x * blockDim.x;
    for (; i < n; i += st) p[i] = 0u;
}

// Wcat[n][k] = k<HID ? W_ih[n][k] : W_hh[n][k-HID]   (bf16);  Wfc bf16; bias=b_ih+b_hh
__global__ void prep_weights(const float* __restrict__ W_ih,
                             const float* __restrict__ W_hh,
                             const float* __restrict__ b_ih,
                             const float* __restrict__ b_hh,
                             const float* __restrict__ W_fc,
                             bf16_t* __restrict__ Wcat,
                             bf16_t* __restrict__ Wfcb,
                             float*  __restrict__ bias)
{
    const size_t nCat = (size_t)GATES4H * KCAT;
    const size_t nFc  = (size_t)VOCAB * HID;
    const size_t tot  = nCat + nFc + GATES4H;
    size_t i = (size_t)blockIdx.x * blockDim.x + threadIdx.x;
    size_t st = (size_t)gridDim.x * blockDim.x;
    for (; i < tot; i += st) {
        if (i < nCat) {
            size_t n = i >> 11, k = i & (KCAT - 1);
            float v = (k < HID) ? W_ih[n * HID + k] : W_hh[n * HID + (k - HID)];
            Wcat[i] = (bf16_t)v;
        } else if (i < nCat + nFc) {
            size_t j = i - nCat;
            Wfcb[j] = (bf16_t)W_fc[j];
        } else {
            size_t j = i - nCat - nFc;
            bias[j] = b_ih[j] + b_hh[j];
        }
    }
}

__global__ void init_state(int* think, int* outs, int* isth, int* anyf, int nany) {
    int i = blockIdx.x * blockDim.x + threadIdx.x;
    if (i < BSZ) { think[i] = 0; outs[i] = 0; isth[i] = 1; }
    if (i < nany) anyf[i] = (i == 0) ? 1 : 0;
}

// A[:,0:HID] <- bf16(x[:, t, :])
__global__ void enc_build_x(const float* __restrict__ x, bf16_t* __restrict__ Abuf, int t) {
    int i = blockIdx.x * blockDim.x + threadIdx.x;   // BSZ*VOCAB threads
    int b = i >> 10, v = i & 1023;
    Abuf[(size_t)b * KCAT + v] = (bf16_t)x[((size_t)b * T_ENC + t) * VOCAB + v];
}

// A[:,0:HID] <- bf16(x[b, lengths[b]-1, :])
__global__ void dec_build_in(const float* __restrict__ x, const int* __restrict__ lengths,
                             bf16_t* __restrict__ Abuf) {
    int i = blockIdx.x * blockDim.x + threadIdx.x;
    int b = i >> 10, v = i & 1023;
    int t = lengths[b] - 1;
    Abuf[(size_t)b * KCAT + v] = (bf16_t)x[((size_t)b * T_ENC + t) * VOCAB + v];
}

// LSTM cell update; also writes bf16(h) into A[:,HID:2*HID] (stride KCAT)
__global__ void lstm_update(const float* __restrict__ gates,
                            float* __restrict__ h, float* __restrict__ c,
                            bf16_t* __restrict__ Ah,   // = Abuf + HID
                            const int* __restrict__ lengths, int t, int masked)
{
    int i = blockIdx.x * blockDim.x + threadIdx.x;   // BSZ*HID threads
    int b = i >> 10, j = i & 1023;
    const float* g = gates + (size_t)b * GATES4H;
    float ig = sig_f(g[j]);
    float fg = sig_f(g[HID + j]);
    float gg = tanhf(g[2 * HID + j]);
    float og = sig_f(g[3 * HID + j]);
    float c_old = c[i], h_old = h[i];
    float cn = fg * c_old + ig * gg;
    float hn = og * tanhf(cn);
    bool upd = masked ? (t < lengths[b]) : true;
    float hv = upd ? hn : h_old;
    float cv = upd ? cn : c_old;
    h[i] = hv; c[i] = cv;
    Ah[(size_t)b * KCAT + j] = (bf16_t)hv;
}

// Decoder control: argmax(logits+gumbel), fo write, next-input build, state update
__global__ __launch_bounds__(256) void dec_ctrl(
    const float* __restrict__ logits, bf16_t* __restrict__ Abuf,
    float* __restrict__ fo,
    int* __restrict__ think, int* __restrict__ outs, int* __restrict__ isth,
    int* __restrict__ anyf, int t)
{
    const int b = blockIdx.x, tid = threadIdx.x;
    __shared__ float smax[256];
    __shared__ int   sidx[256];
    const float* lg = logits + (size_t)b * VOCAB;

    float best = -3.4e38f; int bestv = 0;
    for (int v = tid; v < VOCAB; v += 256) {
        float z = lg[v] + gumbel_f(t, b, v);
        if (z > best) { best = z; bestv = v; }
    }
    smax[tid] = best; sidx[tid] = bestv;
    __syncthreads();
    for (int s = 128; s > 0; s >>= 1) {
        if (tid < s) {
            if (smax[tid + s] > smax[tid] ||
                (smax[tid + s] == smax[tid] && sidx[tid + s] < sidx[tid])) {
                smax[tid] = smax[tid + s]; sidx[tid] = sidx[tid + s];
            }
        }
        __syncthreads();
    }
    const int pred = sidx[0];
    const int it = isth[b], os = outs[b];
    const int any = anyf[t];                         // any(is_think) at step entry
    const int outputting = (!it) && (os < N_OUT);

    if (outputting) {
        float* dst = fo + ((size_t)b * N_OUT + os) * NUM_CLASSES;
        for (int v = tid; v < NUM_CLASSES; v += 256) dst[v] = lg[v];
    }
    bf16_t* arow = Abuf + (size_t)b * KCAT;
    if (any) { for (int v = tid; v < VOCAB; v += 256) arow[v] = (bf16_t)lg[v]; }
    else     { for (int v = tid; v < VOCAB; v += 256) arow[v] = (bf16_t)((v == pred) ? 1.f : 0.f); }

    if (tid == 0) {
        int ts = think[b] + (it ? 1 : 0);
        think[b] = ts;
        outs[b]  = os + (outputting ? 1 : 0);
        int just = it && ((pred == NUM_CLASSES) || (ts >= MAX_THINK));
        int itn  = it && !just;
        isth[b]  = itn;
        if (itn) atomicOr(&anyf[t + 1], 1);
    }
}

__global__ void write_think(const int* __restrict__ think, float* __restrict__ tail) {
    int b = blockIdx.x * blockDim.x + threadIdx.x;
    if (b < BSZ) tail[b] = (float)think[b];
}

// ===========================================================================
// Launch
// ===========================================================================
extern "C" void kernel_launch(void* const* d_in, const int* in_sizes, int n_in,
                              void* d_out, int out_size, void* d_ws, size_t ws_size,
                              hipStream_t stream)
{
    (void)in_sizes; (void)n_in; (void)out_size; (void)ws_size;
    const float* x     = (const float*)d_in[0];
    const float* W_ih  = (const float*)d_in[1];
    const float* W_hh  = (const float*)d_in[2];
    const float* b_ih  = (const float*)d_in[3];
    const float* b_hh  = (const float*)d_in[4];
    const float* W_fc  = (const float*)d_in[5];
    const float* b_fc  = (const float*)d_in[6];
    const int*   lens  = (const int*)d_in[7];

    // workspace carve-out
    char* p = (char*)d_ws;
    auto alloc = [&](size_t bytes) -> void* {
        void* r = (void*)p; p += (bytes + 255) & ~(size_t)255; return r;
    };
    bf16_t* Wcat  = (bf16_t*)alloc((size_t)GATES4H * KCAT * sizeof(bf16_t)); // 16 MB
    bf16_t* Wfcb  = (bf16_t*)alloc((size_t)VOCAB * HID * sizeof(bf16_t));    //  2 MB
    float*  bias  = (float*) alloc((size_t)GATES4H * sizeof(float));
    bf16_t* Abuf  = (bf16_t*)alloc((size_t)BSZ * KCAT * sizeof(bf16_t));     //  1 MB
    float*  gates = (float*) alloc((size_t)BSZ * GATES4H * sizeof(float));   //  4 MB
    float*  hbuf  = (float*) alloc((size_t)BSZ * HID * sizeof(float));
    float*  cbuf  = (float*) alloc((size_t)BSZ * HID * sizeof(float));
    float*  lgts  = (float*) alloc((size_t)BSZ * VOCAB * sizeof(float));
    int*    think = (int*)   alloc(BSZ * sizeof(int));
    int*    outs  = (int*)   alloc(BSZ * sizeof(int));
    int*    isth  = (int*)   alloc(BSZ * sizeof(int));
    int*    anyf  = (int*)   alloc((T_DEC + 1) * sizeof(int));

    float* fo_out    = (float*)d_out;
    float* think_out = (float*)d_out + (size_t)BSZ * N_OUT * NUM_CLASSES;

    // init: weights, zeros, state
    prep_weights<<<1024, 256, 0, stream>>>(W_ih, W_hh, b_ih, b_hh, W_fc, Wcat, Wfcb, bias);
    fill_zero_u32<<<1024, 256, 0, stream>>>((uint32_t*)hbuf, (size_t)BSZ * HID);
    fill_zero_u32<<<1024, 256, 0, stream>>>((uint32_t*)cbuf, (size_t)BSZ * HID);
    fill_zero_u32<<<1024, 256, 0, stream>>>((uint32_t*)Abuf, (size_t)BSZ * KCAT / 2); // bf16 zeros
    fill_zero_u32<<<4096, 256, 0, stream>>>((uint32_t*)fo_out,
                                            (size_t)BSZ * N_OUT * NUM_CLASSES + BSZ);
    init_state<<<2, 256, 0, stream>>>(think, outs, isth, anyf, T_DEC + 1);

    const dim3 gGates(GATES4H / 512, BSZ / 32);   // (8,8)
    const dim3 gLogit(VOCAB   / 512, BSZ / 32);   // (2,8)
    const int  nElem = (BSZ * HID) / 256;         // 1024 blocks for elementwise

    // ---------------- encoder ----------------
    for (int t = 0; t < T_ENC; ++t) {
        enc_build_x<<<nElem, 256, 0, stream>>>(x, Abuf, t);
        gemm_bf16_wmma<<<gGates, 256, 0, stream>>>(Abuf, KCAT, Wcat, KCAT, bias,
                                                   gates, GATES4H, KCAT);
        lstm_update<<<nElem, 256, 0, stream>>>(gates, hbuf, cbuf, Abuf + HID,
                                               lens, t, 1);
    }

    // ---------------- decoder ----------------
    dec_build_in<<<nElem, 256, 0, stream>>>(x, lens, Abuf);
    for (int t = 0; t < T_DEC; ++t) {
        gemm_bf16_wmma<<<gGates, 256, 0, stream>>>(Abuf, KCAT, Wcat, KCAT, bias,
                                                   gates, GATES4H, KCAT);
        lstm_update<<<nElem, 256, 0, stream>>>(gates, hbuf, cbuf, Abuf + HID,
                                               lens, t, 0);
        gemm_bf16_wmma<<<gLogit, 256, 0, stream>>>(Abuf + HID, KCAT, Wfcb, HID, b_fc,
                                                   lgts, VOCAB, HID);
        dec_ctrl<<<BSZ, 256, 0, stream>>>(lgts, Abuf, fo_out, think, outs, isth, anyf, t);
    }
    write_think<<<1, 256, 0, stream>>>(think, think_out);
}
